// ConvAddAttention_27745488732211
// MI455X (gfx1250) — matrix operand order repaired and checked by reading, not compiled, not run on hardware
//
#include <hip/hip_runtime.h>
#include <hip/hip_bf16.h>

// ---------------------------------------------------------------------------
// Problem constants: x (8,512,32,32) fp32 -> GEMM X(8192x512) * proj_w^T,
// score GEMV, per-5x5-window LN->softmax->weighted sum, output (8,512,28,28).
// ---------------------------------------------------------------------------
#define NB    8
#define CH    512
#define HH    32
#define WW    32
#define OCH   512
#define KW    5
#define HOUT  28
#define WOUT  28
#define NPIX  (NB * HH * WW)          // 8192 GEMM rows
#define KSTEPS (CH / 32)              // 16 k-steps of 32
#define MTILES (NPIX / 16)            // 512
#define NTILES (OCH / 16)             // 32
#define EPSV  1e-05f

typedef __attribute__((ext_vector_type(16))) _Float16 v16h;
typedef __attribute__((ext_vector_type(8)))  float    v8f;

// Packed-fragment layout: half-index = ((tile*KSTEPS + ks)*32 + lane)*32 + j
//   j = 0..15 -> hi fragment halves, j = 16..31 -> lo (residual) halves.
// 64 B per (lane, k-step); consecutive lanes contiguous -> coalesced.
#define FRAG_BASE(tile, ks, lane) ((((size_t)(tile) * KSTEPS + (ks)) * 32 + (lane)) * 32)

// ---------------------------------------------------------------------------
// Wave32 reductions
// ---------------------------------------------------------------------------
__device__ __forceinline__ float wred_sum(float v) {
#pragma unroll
    for (int off = 16; off > 0; off >>= 1) v += __shfl_xor(v, off, 32);
    return v;
}
__device__ __forceinline__ float wred_max(float v) {
#pragma unroll
    for (int off = 16; off > 0; off >>= 1) v = fmaxf(v, __shfl_xor(v, off, 32));
    return v;
}

__device__ __forceinline__ void split16(const float* __restrict__ f, v16h& hi, v16h& lo) {
#pragma unroll
    for (int i = 0; i < 16; ++i) {
        _Float16 h = (_Float16)f[i];
        hi[i] = h;
        lo[i] = (_Float16)(f[i] - (float)h);
    }
}

// ---------------------------------------------------------------------------
// One-shot conversion: X -> packed A fragments (hi/lo f16), WMMA lane order.
// ---------------------------------------------------------------------------
__global__ __launch_bounds__(256) void k_cvtA(const float* __restrict__ X,
                                              _Float16* __restrict__ Apk) {
    const int gid  = blockIdx.x * 256 + threadIdx.x;   // 512*16*32 = 262144
    const int tile = gid >> 9;
    const int rem  = gid & 511;
    const int ks   = rem >> 5;
    const int lane = rem & 31;
    const int lm = lane & 15, lk = lane >> 4;

    const float* ap = X + (size_t)(tile * 16 + lm) * CH + ks * 32 + lk * 8;
    float f[16];
    *(float4*)&f[0]  = *(const float4*)(ap + 0);
    *(float4*)&f[4]  = *(const float4*)(ap + 4);
    *(float4*)&f[8]  = *(const float4*)(ap + 16);
    *(float4*)&f[12] = *(const float4*)(ap + 20);

    v16h hi, lo;
    split16(f, hi, lo);
    _Float16* dst = Apk + FRAG_BASE(tile, ks, lane);
    *(v16h*)(dst)      = hi;
    *(v16h*)(dst + 16) = lo;
}

// ---------------------------------------------------------------------------
// One-shot conversion: proj_w -> packed B fragments (hi/lo f16).
// B[k][n] = proj_w[n][k]; contiguous along C, no transpose needed.
// ---------------------------------------------------------------------------
__global__ __launch_bounds__(256) void k_cvtB(const float* __restrict__ Wp,
                                              _Float16* __restrict__ Bpk) {
    const int gid  = blockIdx.x * 256 + threadIdx.x;   // 32*16*32 = 16384
    const int tile = gid >> 9;
    const int rem  = gid & 511;
    const int ks   = rem >> 5;
    const int lane = rem & 31;
    const int lm = lane & 15, lk = lane >> 4;

    const float* bp = Wp + (size_t)(tile * 16 + lm) * CH + ks * 32 + lk * 8;
    float f[16];
    *(float4*)&f[0]  = *(const float4*)(bp + 0);
    *(float4*)&f[4]  = *(const float4*)(bp + 4);
    *(float4*)&f[8]  = *(const float4*)(bp + 16);
    *(float4*)&f[12] = *(const float4*)(bp + 20);

    v16h hi, lo;
    split16(f, hi, lo);
    _Float16* dst = Bpk + FRAG_BASE(tile, ks, lane);
    *(v16h*)(dst)      = hi;
    *(v16h*)(dst + 16) = lo;
}

// ---------------------------------------------------------------------------
// score GEMV: score[p] = dot(X[p,:], score_w) + score_b
// ---------------------------------------------------------------------------
__global__ __launch_bounds__(256) void k_score(const float* __restrict__ X,
                                               const float* __restrict__ sw,
                                               const float* __restrict__ sb,
                                               float* __restrict__ score) {
    const int lane = threadIdx.x & 31;
    const int wave = threadIdx.x >> 5;
    const int row  = blockIdx.x * 8 + wave;
    const float* xr = X + (size_t)row * CH;

    float acc = 0.f;
#pragma unroll
    for (int q = 0; q < 4; ++q) {
        const int idx = q * 128 + lane * 4;
        float4 xv = *(const float4*)(xr + idx);
        float4 wv = *(const float4*)(sw + idx);
        acc = fmaf(xv.x, wv.x, acc);
        acc = fmaf(xv.y, wv.y, acc);
        acc = fmaf(xv.z, wv.z, acc);
        acc = fmaf(xv.w, wv.w, acc);
    }
    acc = wred_sum(acc);
    if (lane == 0) score[row] = acc + sb[0];
}

// ---------------------------------------------------------------------------
// WMMA GEMM: proj = X * proj_w^T + proj_b on pre-packed split-f16 fragments.
// Block: 8 waves = 128(M) x 64(O); wave: 32x32 (2x2 subtiles).
// B fragments staged into LDS via GLOBAL_LOAD_ASYNC_TO_LDS_B128 (ASYNCcnt),
// in two 64 KB phases (8 k-steps each); A streams global->VGPR concurrently.
// Inner loop: 8 global b128 (A) + 8 ds b128 (B) + 12 v_wmma_f32_16x16x32_f16.
// ---------------------------------------------------------------------------
__global__ __launch_bounds__(256) void k_gemm(const _Float16* __restrict__ Apk,
                                              const _Float16* __restrict__ Bpk,
                                              const float* __restrict__ Bp,
                                              float* __restrict__ proj) {
    // 4 N-tiles x 8 k-steps x 32 lanes x 32 halves = 32768 halves = 64 KB
    __shared__ __align__(128) _Float16 Bs[4 * 8 * 32 * 32];

    const int tid  = threadIdx.x;
    const int bm   = blockIdx.x >> 3;     // 0..63
    const int bn   = blockIdx.x & 7;      // 0..7
    const int wave = tid >> 5;
    const int lane = tid & 31;
    const int wm   = wave & 3;
    const int wn   = wave >> 2;

    const int mt0 = bm * 8 + wm * 2;      // first of 2 M tiles (16 rows each)
    const int lm = lane & 15, lk = lane >> 4;

    const uint32_t ldsbase = (uint32_t)(uintptr_t)Bs;
    v8f acc[2][2] = {};

#pragma unroll
    for (int phase = 0; phase < 2; ++phase) {
        if (phase) __syncthreads();       // phase-0 readers done before overwrite

        // ---- async stage: 64 KB of B fragments (4 tiles, 8 k-steps) -> LDS
#pragma unroll
        for (int i = 0; i < 16; ++i) {
            const int idx = tid + i * 256;        // 4096 x 16 B chunks
            const int c   = idx >> 10;            // N-tile chunk (16 KB each)
            const int w   = idx & 1023;           // 16 B unit within chunk
            const uint32_t lb = ldsbase + (c << 14) + (w << 4);
            const uint64_t ga = (uint64_t)(uintptr_t)Bpk +
                                (((size_t)(bn * 4 + c) * KSTEPS + phase * 8) << 11) +
                                ((size_t)w << 4);
            asm volatile("global_load_async_to_lds_b128 %0, %1, off"
                         :: "v"(lb), "v"(ga) : "memory");
        }
        asm volatile("s_wait_asynccnt 0x0" ::: "memory");
        __syncthreads();

        // ---- 8 k-steps on staged data
        for (int kss = 0; kss < 8; ++kss) {
            const int ks = phase * 8 + kss;

            v16h ah[2], al[2], bh[2], bl[2];
#pragma unroll
            for (int s = 0; s < 2; ++s) {
                const _Float16* p = Apk + FRAG_BASE(mt0 + s, ks, lane);
                ah[s] = *(const v16h*)(p);
                al[s] = *(const v16h*)(p + 16);
            }
#pragma unroll
            for (int u = 0; u < 2; ++u) {
                const int ut = wn * 2 + u;        // local N tile 0..3
                const _Float16* p = &Bs[((ut * 8 + kss) * 32 + lane) * 32];
                bh[u] = *(const v16h*)(p);
                bl[u] = *(const v16h*)(p + 16);
            }
            // term-major: same-accumulator WMMAs spaced 4 apart
#pragma unroll
            for (int s = 0; s < 2; ++s)
#pragma unroll
                for (int u = 0; u < 2; ++u)
                    acc[s][u] = __builtin_amdgcn_wmma_f32_16x16x32_f16(
                        false, ah[s], false, bh[u], (short)0, acc[s][u], false, false);
#pragma unroll
            for (int s = 0; s < 2; ++s)
#pragma unroll
                for (int u = 0; u < 2; ++u)
                    acc[s][u] = __builtin_amdgcn_wmma_f32_16x16x32_f16(
                        false, ah[s], false, bl[u], (short)0, acc[s][u], false, false);
#pragma unroll
            for (int s = 0; s < 2; ++s)
#pragma unroll
                for (int u = 0; u < 2; ++u)
                    acc[s][u] = __builtin_amdgcn_wmma_f32_16x16x32_f16(
                        false, al[s], false, bh[u], (short)0, acc[s][u], false, false);
        }
    }

    // D layout: VGPR r -> M = 8*(lane/16)+r, N = lane%16 ; add bias.
#pragma unroll
    for (int u = 0; u < 2; ++u) {
        const int col = (bn * 4 + wn * 2 + u) * 16 + lm;
        const float bias = Bp[col];
#pragma unroll
        for (int s = 0; s < 2; ++s) {
#pragma unroll
            for (int r = 0; r < 8; ++r) {
                const int row = (mt0 + s) * 16 + lk * 8 + r;
                proj[(size_t)row * OCH + col] = acc[s][u][r] + bias;
            }
        }
    }
}

// ---------------------------------------------------------------------------
// Per-window LayerNorm(25 scores) -> softmax -> weighted sum of 25 proj rows.
// ---------------------------------------------------------------------------
__global__ __launch_bounds__(256) void k_attn(const float* __restrict__ proj,
                                              const float* __restrict__ score,
                                              const float* __restrict__ lnw,
                                              const float* __restrict__ lnb,
                                              float* __restrict__ out) {
    __shared__ float wv[KW * KW];

    const int bid = blockIdx.x;
    const int n   = bid / (HOUT * WOUT);
    const int rem = bid - n * (HOUT * WOUT);
    const int i   = rem / WOUT;
    const int j   = rem - i * WOUT;
    const int tid = threadIdx.x;

    if (tid < 32) {
        const int  t      = tid;
        const bool active = (t < KW * KW);
        const int  tr = t / KW, tc = t - KW * (t / KW);
        float sc = 0.f;
        if (active) sc = score[(n * HH + (i + tr)) * WW + (j + tc)];
        const float sum   = wred_sum(sc);
        const float sumsq = wred_sum(sc * sc);
        const float mu    = sum * (1.0f / 25.0f);
        const float var   = sumsq * (1.0f / 25.0f) - mu * mu;
        const float inv   = rsqrtf(var + EPSV);
        float scn = active ? (sc - mu) * inv * lnw[t] + lnb[t] : -1e30f;
        const float m = wred_max(scn);
        const float e = active ? __expf(scn - m) : 0.f;
        const float s = wred_sum(e);
        if (active) wv[t] = e / s;
    }
    __syncthreads();

    int pb[KW * KW];
#pragma unroll
    for (int t = 0; t < KW * KW; ++t) {
        const int tr = t / KW, tc = t - KW * (t / KW);
        pb[t] = (n * HH + (i + tr)) * WW + (j + tc);
    }

#pragma unroll
    for (int rep = 0; rep < 2; ++rep) {
        const int o = tid + rep * 256;
        float acc = 0.f;
#pragma unroll
        for (int t = 0; t < KW * KW; ++t)
            acc = fmaf(wv[t], proj[(size_t)pb[t] * OCH + o], acc);
        out[((size_t)(n * OCH + o) * HOUT + i) * WOUT + j] = acc;
    }
}

// ---------------------------------------------------------------------------
// Launch. Workspace: proj (16 MB) | score (32 KB) | Apk (16 MB) | Bpk (1 MB).
// ---------------------------------------------------------------------------
extern "C" void kernel_launch(void* const* d_in, const int* in_sizes, int n_in,
                              void* d_out, int out_size, void* d_ws, size_t ws_size,
                              hipStream_t stream) {
    const float* x       = (const float*)d_in[0];
    const float* proj_w  = (const float*)d_in[1];
    const float* proj_b  = (const float*)d_in[2];
    const float* score_w = (const float*)d_in[3];
    const float* score_b = (const float*)d_in[4];
    const float* ln_w    = (const float*)d_in[5];
    const float* ln_b    = (const float*)d_in[6];

    char* ws = (char*)d_ws;
    float*    proj  = (float*)ws;    ws += (size_t)NPIX * OCH * sizeof(float);
    float*    score = (float*)ws;    ws += (size_t)NPIX * sizeof(float);
    _Float16* Apk   = (_Float16*)ws; ws += (size_t)MTILES * KSTEPS * 32 * 32 * sizeof(_Float16);
    _Float16* Bpk   = (_Float16*)ws;

    k_cvtA <<<MTILES * KSTEPS * 32 / 256, 256, 0, stream>>>(x, Apk);
    k_cvtB <<<NTILES * KSTEPS * 32 / 256, 256, 0, stream>>>(proj_w, Bpk);
    k_score<<<NPIX / 8, 256, 0, stream>>>(x, score_w, score_b, score);
    k_gemm <<<(NPIX / 128) * (OCH / 64), 256, 0, stream>>>(Apk, Bpk, proj_b, proj);
    k_attn <<<NB * HOUT * WOUT, 256, 0, stream>>>(proj, score, ln_w, ln_b, (float*)d_out);
}
